// SelfAttention_996432413368
// MI455X (gfx1250) — matrix operand order repaired and checked by reading, not compiled, not run on hardware
//
#include <hip/hip_runtime.h>

// ---------------------------------------------------------------------------
// Types
// ---------------------------------------------------------------------------
typedef unsigned short u16;
typedef __attribute__((ext_vector_type(16))) __bf16 v16bf;
typedef __attribute__((ext_vector_type(8)))  float  v8f;

#define NB   2
#define NH   16
#define SEQ  2048
#define DIM  2048
#define HD   128            // head dim
#define QKVN 6144           // 3*DIM

// f32 -> bf16 round-to-nearest-even
__device__ __forceinline__ u16 f2bf(float f) {
  unsigned int u = __float_as_uint(f);
  u += 0x7FFFu + ((u >> 16) & 1u);
  return (u16)(u >> 16);
}
__device__ __forceinline__ float bf2f(u16 h) {
  return __uint_as_float(((unsigned int)h) << 16);
}

// ---------------------------------------------------------------------------
// CDNA5 async global->LDS copy (ASYNCcnt-tracked), inline asm per ISA 10.x.
// GVS mode: mem_addr = SADDR(64b) + VADDR(32b offset).  VDST carries the LDS
// byte address (low 32 bits of the generic shared pointer).
// ---------------------------------------------------------------------------
__device__ __forceinline__ void async_copy_b128(void* lds_ptr,
                                                const void* gbase,
                                                unsigned byte_off) {
  unsigned lds_addr = (unsigned)(unsigned long long)lds_ptr;
  asm volatile("global_load_async_to_lds_b128 %0, %1, %2"
               :: "v"(lds_addr), "v"(byte_off), "s"(gbase)
               : "memory");
}
__device__ __forceinline__ void wait_async0() {
  asm volatile("s_wait_asynccnt 0" ::: "memory");
}

// ---------------------------------------------------------------------------
// WMMA fragment helpers (CDNA5 16x16x32 bf16, wave32 layouts per ISA 7.12.2)
// ---------------------------------------------------------------------------
struct Frag {
  union { v16bf v; uint4 q[2]; };
};

// A matrix 16x32: lane row = lane%16; halves 0..7 -> K=(lane/16)*8 + e,
// halves 8..15 -> K=16+(lane/16)*8 + (e-8).  Two aligned 16B LDS reads.
__device__ __forceinline__ Frag load_frag_a(const u16* base, int ld, int lane) {
  Frag f;
  const u16* row = base + (lane & 15) * ld + ((lane >> 4) << 3);
  f.q[0] = *(const uint4*)(row);
  f.q[1] = *(const uint4*)(row + 16);
  return f;
}

// B matrix 32x16: lane col = lane%16; halves 0..7 -> K=(lane/16)*8 + e,
// halves 8..15 -> K=16+(lane/16)*8 + (e-8).
__device__ __forceinline__ Frag load_frag_b(const u16* base, int ld, int lane) {
  Frag f;
  const u16* row = base + (lane & 15) * ld + ((lane >> 4) << 3);
  f.q[0] = *(const uint4*)(row);
  f.q[1] = *(const uint4*)(row + 16);
  return f;
}

__device__ __forceinline__ v8f wmma_bf16(v16bf a, v16bf b, v8f c) {
  return __builtin_amdgcn_wmma_f32_16x16x32_bf16(
      /*neg_a=*/false, a, /*neg_b=*/false, b,
      /*c_mod=*/(short)0, c, /*reuse_a=*/false, /*reuse_b=*/false);
}

// ---------------------------------------------------------------------------
// Kernel 0: fp32 -> bf16 elementwise convert (vectorized x4).
// ---------------------------------------------------------------------------
__global__ __launch_bounds__(256) void cvt_bf16_kernel(
    const float* __restrict__ src, u16* __restrict__ dst) {
  long i = ((long)blockIdx.x * 256 + threadIdx.x) * 4;
  float4 f = *(const float4*)(src + i);
  ushort4 h;
  h.x = f2bf(f.x); h.y = f2bf(f.y); h.z = f2bf(f.z); h.w = f2bf(f.w);
  *(ushort4*)(dst + i) = h;
}

// ---------------------------------------------------------------------------
// Kernel 1: QKV projection GEMM  C[4096,6144] = X[4096,2048] @ W[6144,2048]^T+b
// bf16 inputs staged via async global->LDS copies, double-buffered (K-slab 64).
// Epilogue scatters into q/k/v bf16 buffers laid out [b, h, n, d].
// Channel layout (Flux 'b n (h d j)'): c = head*384 + dd*3 + j.
// Block tile 128x128, 8 waves each computing 32x64.
// ---------------------------------------------------------------------------
__global__ __launch_bounds__(256) void qkv_gemm_kernel(
    const u16* __restrict__ xh, const u16* __restrict__ wh,
    const float* __restrict__ bias,
    u16* __restrict__ qB, u16* __restrict__ kB, u16* __restrict__ vB) {
  __shared__ u16 ldsA[2][128 * 72];   // 128 rows x 64 k (+8 pad), double buffer
  __shared__ u16 ldsB[2][128 * 72];

  const int tid  = threadIdx.x;
  const int lane = tid & 31;
  const int wv   = tid >> 5;
  const int wm   = wv & 3;    // 4 wave rows * 32 = 128
  const int wn   = wv >> 2;   // 2 wave cols * 64 = 128
  const int gm0  = blockIdx.y * 128;
  const int gn0  = blockIdx.x * 128;

  const v8f vzero = {};
  v8f acc[2][4];
  for (int mi = 0; mi < 2; ++mi)
    for (int ni = 0; ni < 4; ++ni) acc[mi][ni] = vzero;

  // Per-thread staging offsets (loop-invariant): 1024 16B chunks per matrix.
  const int srow = tid >> 3;          // chunk row  (two rows per 16 threads? no: 8 chunks/row)
  const int sc8  = tid & 7;           // chunk col within row (8 halfs each)

  // Stage one 128x64 A/B slab into the given LDS buffers (async, no VGPRs).
  auto stage = [&](u16* abuf, u16* bbuf, int k0) {
    for (int i = 0; i < 4; ++i) {
      int row = srow + i * 32;        // 4 x 32 rows
      unsigned goA = (unsigned)(((gm0 + row) * DIM + k0 + sc8 * 8) * 2);
      unsigned goB = (unsigned)(((gn0 + row) * DIM + k0 + sc8 * 8) * 2);
      async_copy_b128(abuf + row * 72 + sc8 * 8, xh, goA);
      async_copy_b128(bbuf + row * 72 + sc8 * 8, wh, goB);
    }
  };

  u16* aCur = ldsA[0]; u16* aNxt = ldsA[1];
  u16* bCur = ldsB[0]; u16* bNxt = ldsB[1];

  stage(aCur, bCur, 0);
  wait_async0();
  __syncthreads();

#pragma unroll 1
  for (int kt = 0; kt < DIM / 64; ++kt) {
    if (kt + 1 < DIM / 64) stage(aNxt, bNxt, (kt + 1) * 64);

    for (int kc = 0; kc < 64; kc += 32) {
      Frag fa[2], fb[4];
      for (int mi = 0; mi < 2; ++mi)
        fa[mi] = load_frag_a(aCur + (wm * 32 + mi * 16) * 72 + kc, 72, lane);
      for (int ni = 0; ni < 4; ++ni)
        fb[ni] = load_frag_b(bCur + (wn * 64 + ni * 16) * 72 + kc, 72, lane);
      for (int mi = 0; mi < 2; ++mi)
        for (int ni = 0; ni < 4; ++ni)
          acc[mi][ni] = wmma_bf16(fa[mi].v, fb[ni].v, acc[mi][ni]);
    }

    wait_async0();
    __syncthreads();
    u16* t0 = aCur; aCur = aNxt; aNxt = t0;
    u16* t1 = bCur; bCur = bNxt; bNxt = t1;
  }

  // Epilogue: bias + scatter into q/k/v [b,h,n,d] bf16.
  for (int ni = 0; ni < 4; ++ni) {
    int c    = gn0 + wn * 64 + ni * 16 + (lane & 15);
    float bc = bias[c];
    int head = c / 384;
    int rr   = c - head * 384;
    int dd   = rr / 3;
    int j    = rr - dd * 3;
    u16* dst = (j == 0) ? qB : (j == 1) ? kB : vB;
    for (int mi = 0; mi < 2; ++mi) {
      v8f a = acc[mi][ni];
      for (int r = 0; r < 8; ++r) {
        int m  = gm0 + wm * 32 + mi * 16 + ((lane >> 4) << 3) + r;
        int bi = m >> 11;
        int nn = m & (SEQ - 1);
        dst[(((long)(bi * NH + head) * SEQ + nn) << 7) + dd] = f2bf(a[r] + bc);
      }
    }
  }
}

// ---------------------------------------------------------------------------
// Kernel 2: RMSNorm + RoPE in place on a bf16 [b,h,n,d] buffer.
// One wave per row; lane owns 4 contiguous elements (= 2 rope pairs).
// rot layout: [n, 64, 2, 2] fp32.
// ---------------------------------------------------------------------------
__global__ __launch_bounds__(256) void norm_rope_kernel(
    u16* __restrict__ t, const float* __restrict__ nw,
    const float* __restrict__ rot) {
  const int lane = threadIdx.x & 31;
  const int wv   = threadIdx.x >> 5;
  const long row = (long)blockIdx.x * 8 + wv;       // 0 .. b*h*n-1
  const int  n   = (int)(row & (SEQ - 1));
  u16* p = t + (row << 7) + lane * 4;

  float v0 = bf2f(p[0]), v1 = bf2f(p[1]), v2 = bf2f(p[2]), v3 = bf2f(p[3]);
  float ss = v0 * v0 + v1 * v1 + v2 * v2 + v3 * v3;
  for (int off = 1; off < 32; off <<= 1) ss += __shfl_xor(ss, off, 32);
  float inv = rsqrtf(ss * (1.0f / HD) + 1e-6f);

  v0 *= inv * nw[lane * 4 + 0];
  v1 *= inv * nw[lane * 4 + 1];
  v2 *= inv * nw[lane * 4 + 2];
  v3 *= inv * nw[lane * 4 + 3];

  const float* r0 = rot + ((long)n * 64 + lane * 2) * 4;
  float o0 = r0[0] * v0 + r0[1] * v1;
  float o1 = r0[2] * v0 + r0[3] * v1;
  float o2 = r0[4] * v2 + r0[5] * v3;
  float o3 = r0[6] * v2 + r0[7] * v3;

  p[0] = f2bf(o0); p[1] = f2bf(o1); p[2] = f2bf(o2); p[3] = f2bf(o3);
}

// ---------------------------------------------------------------------------
// Kernel 3: flash attention, one (b, h, 128-row q tile) per block.
// 8 waves x 16 q-rows. KV tiles of 64 rows. Online softmax in fp32.
// Q and K tiles staged with async global->LDS copies; V transposed manually.
// ---------------------------------------------------------------------------
__global__ __launch_bounds__(256) void flash_attn_kernel(
    const u16* __restrict__ qB, const u16* __restrict__ kB,
    const u16* __restrict__ vB, float* __restrict__ out) {
  __shared__ u16 Qs[128 * 136];     // 128 x 128 (+8 pad)
  __shared__ u16 Ks[64 * 136];      // 64 x 128  (+8 pad)
  __shared__ u16 Vt[128 * 72];      // transposed: Vt[d][kv] (+8 pad)
  __shared__ u16 Ps[8 * 16 * 72];   // per-wave P scratch 16 x 64 (+8 pad)

  const int tid  = threadIdx.x;
  const int lane = tid & 31;
  const int wv   = tid >> 5;
  const int q0   = blockIdx.x * 128;
  const int h    = blockIdx.y;
  const int bi   = blockIdx.z;
  const long bh  = (long)(bi * NH + h);

  const u16* Qg = qB + ((bh * SEQ + q0) << 7);
  for (int i = 0; i < 8; ++i) {                 // 2048 16B chunks / 256 thr
    int fi = tid + i * 256;
    int row = fi >> 4, c = fi & 15;
    async_copy_b128(&Qs[row * 136 + c * 8], Qg,
                    (unsigned)((row * HD + c * 8) * 2));
  }

  const v8f vzero = {};
  v8f o[8];
  float m_i[8], l_i[8];
  for (int f = 0; f < 8; ++f) o[f] = vzero;
  for (int r = 0; r < 8; ++r) { m_i[r] = -1e30f; l_i[r] = 0.0f; }
  u16* Pw = Ps + wv * 16 * 72;
  const float scale = 0.088388347648318447f;    // 1/sqrt(128)

#pragma unroll 1
  for (int kt = 0; kt < SEQ / 64; ++kt) {
    const int k0 = kt * 64;
    const u16* Kg = kB + ((bh * SEQ + k0) << 7);
    const u16* Vg = vB + ((bh * SEQ + k0) << 7);
    __syncthreads();                            // previous tile fully consumed
    for (int i = 0; i < 4; ++i) {               // 1024 16B chunks / 256 thr
      int fi = tid + i * 256;
      int row = fi >> 4, c = fi & 15;
      async_copy_b128(&Ks[row * 136 + c * 8], Kg,
                      (unsigned)((row * HD + c * 8) * 2));
      union { uint4 q; u16 s[8]; } vv;
      vv.q = *(const uint4*)(Vg + (long)row * HD + c * 8);
      for (int j = 0; j < 8; ++j) Vt[(c * 8 + j) * 72 + row] = vv.s[j];
    }
    wait_async0();
    __syncthreads();

    // S = Q_w[16,128] @ K^T[128,64]
    v8f s[4];
    for (int nf = 0; nf < 4; ++nf) s[nf] = vzero;
    for (int kc = 0; kc < HD; kc += 32) {
      Frag aq = load_frag_a(&Qs[(wv * 16) * 136 + kc], 136, lane);
      for (int nf = 0; nf < 4; ++nf) {
        Frag bk = load_frag_b(&Ks[(nf * 16) * 136 + kc], 136, lane);
        s[nf] = wmma_bf16(aq.v, bk.v, s[nf]);
      }
    }

    // Online softmax (row stats replicated across the 16-lane column group).
    float corr[8];
    for (int r = 0; r < 8; ++r) {
      float mc = -1e30f;
      for (int nf = 0; nf < 4; ++nf) {
        float sv = s[nf][r] * scale;
        s[nf][r] = sv;
        mc = fmaxf(mc, sv);
      }
      for (int off = 1; off < 16; off <<= 1) mc = fmaxf(mc, __shfl_xor(mc, off, 32));
      float mn = fmaxf(m_i[r], mc);
      corr[r]  = __expf(m_i[r] - mn);
      m_i[r]   = mn;
      float rs = 0.0f;
      for (int nf = 0; nf < 4; ++nf) {
        float pv = __expf(s[nf][r] - mn);
        rs += pv;
        Pw[(((lane >> 4) << 3) + r) * 72 + nf * 16 + (lane & 15)] = f2bf(pv);
      }
      for (int off = 1; off < 16; off <<= 1) rs += __shfl_xor(rs, off, 32);
      l_i[r] = l_i[r] * corr[r] + rs;
    }
    for (int f = 0; f < 8; ++f)
      for (int r = 0; r < 8; ++r) o[f][r] *= corr[r];

    // O += P[16,64] @ V[64,128]  (P re-enters via LDS as A fragments)
    for (int kc2 = 0; kc2 < 2; ++kc2) {
      Frag ap = load_frag_a(Pw + kc2 * 32, 72, lane);
      for (int nf = 0; nf < 8; ++nf) {
        Frag bv = load_frag_b(&Vt[(nf * 16) * 72 + kc2 * 32], 72, lane);
        o[nf] = wmma_bf16(ap.v, bv.v, o[nf]);
      }
    }
  }

  // Normalize and write fp32 output [b, n, h*d].
  for (int f = 0; f < 8; ++f) {
    for (int r = 0; r < 8; ++r) {
      int row = q0 + wv * 16 + ((lane >> 4) << 3) + r;
      int col = h * HD + f * 16 + (lane & 15);
      out[((long)(bi * SEQ + row)) * DIM + col] = o[f][r] / l_i[r];
    }
  }
}

// ---------------------------------------------------------------------------
// Launch
// ---------------------------------------------------------------------------
extern "C" void kernel_launch(void* const* d_in, const int* in_sizes, int n_in,
                              void* d_out, int out_size, void* d_ws, size_t ws_size,
                              hipStream_t stream) {
  const float* x    = (const float*)d_in[0];
  const float* w    = (const float*)d_in[1];
  const float* bias = (const float*)d_in[2];
  const float* qw   = (const float*)d_in[3];
  const float* kw   = (const float*)d_in[4];
  const float* rot  = (const float*)d_in[5];
  float* out        = (float*)d_out;

  const long nx  = (long)NB * SEQ * DIM;       // 8,388,608
  const long nw_ = (long)QKVN * DIM;           // 12,582,912
  const long per = (long)NB * NH * SEQ * HD;   // 8,388,608

  u16* xh = (u16*)d_ws;
  u16* wh = xh + nx;
  u16* qB = wh + nw_;
  u16* kB = qB + per;
  u16* vB = kB + per;

  cvt_bf16_kernel<<<(int)(nx / 4 / 256), 256, 0, stream>>>(x, xh);
  cvt_bf16_kernel<<<(int)(nw_ / 4 / 256), 256, 0, stream>>>(w, wh);
  qkv_gemm_kernel<<<dim3(QKVN / 128, (NB * SEQ) / 128), 256, 0, stream>>>(
      xh, wh, bias, qB, kB, vB);
  norm_rope_kernel<<<(NB * NH * SEQ) / 8, 256, 0, stream>>>(qB, qw, rot);
  norm_rope_kernel<<<(NB * NH * SEQ) / 8, 256, 0, stream>>>(kB, kw, rot);
  flash_attn_kernel<<<dim3(SEQ / 128, NH, NB), 256, 0, stream>>>(qB, kB, vB, out);
}
